// PTransformerBlock_47571057771034
// MI455X (gfx1250) — matrix operand order repaired and checked by reading, not compiled, not run on hardware
//
#include <hip/hip_runtime.h>
#include <hip/hip_bf16.h>

// ---------------------------------------------------------------------------
// Transformer block for MI455X (gfx1250, wave32, WMMA).
// bf16 storage / f32 accumulate via v_wmma_f32_16x16x32_bf16.
// GEMM uses a double-buffered async global->LDS pipeline (ASYNCcnt-tracked).
// ---------------------------------------------------------------------------

typedef __attribute__((ext_vector_type(16))) __bf16        v16bf;
typedef __attribute__((ext_vector_type(8)))  float         v8f;
typedef __attribute__((ext_vector_type(8)))  unsigned int  v8u;
typedef int v4i __attribute__((vector_size(16)));          // matches async builtin

#define SQ   2048
#define BB   4
#define DD   1024
#define HH   16
#define KK   64
#define FFD  4096
#define MROWS (SQ * BB)          // 8192 token rows

#if defined(__AMDGCN__) && __has_builtin(__builtin_amdgcn_global_load_async_to_lds_b128)
#define ASYNC_LDS 1
#else
#define ASYNC_LDS 0
#endif

// ---- helpers ---------------------------------------------------------------

__device__ __forceinline__ unsigned short f2bf(float f) {
    unsigned int u = __builtin_bit_cast(unsigned int, f);
    u += 0x7FFFu + ((u >> 16) & 1u);           // round-to-nearest-even
    return (unsigned short)(u >> 16);
}

// Build a 16-element bf16 fragment from two 8-bf16 (16B) chunks.
__device__ __forceinline__ v16bf ld_frag16(const unsigned short* p0,
                                           const unsigned short* p1) {
    uint4 a = *(const uint4*)p0;
    uint4 b = *(const uint4*)p1;
    v8u u;
    u[0] = a.x; u[1] = a.y; u[2] = a.z; u[3] = a.w;
    u[4] = b.x; u[5] = b.y; u[6] = b.z; u[7] = b.w;
    return __builtin_bit_cast(v16bf, u);
}

__device__ __forceinline__ v8f wmma_bf16(v16bf a, v16bf b, v8f c) {
    return __builtin_amdgcn_wmma_f32_16x16x32_bf16(
        false, a, false, b, (short)0, c, false, false);
}

#if ASYNC_LDS
// Async copy of 16 bytes (8 bf16) from global to LDS, tracked by ASYNCcnt.
__device__ __forceinline__ void gld_async16(const unsigned short* g,
                                            unsigned short* l) {
    __builtin_amdgcn_global_load_async_to_lds_b128(
        (__attribute__((address_space(1))) v4i*)g,
        (__attribute__((address_space(3))) v4i*)l, 0, 0);
}

template <int N>
__device__ __forceinline__ void wait_async() {
#if __has_builtin(__builtin_amdgcn_s_wait_asynccnt)
    __builtin_amdgcn_s_wait_asynccnt(N);
#else
    asm volatile("s_wait_asynccnt %0" :: "n"(N) : "memory");
#endif
}
#endif

// ---- f32 -> bf16 conversion (n must be a multiple of 1024) -----------------

__global__ __launch_bounds__(256) void cvt_bf16_kernel(
        const float* __restrict__ in, unsigned short* __restrict__ out) {
    const size_t i = ((size_t)blockIdx.x * 256 + threadIdx.x) * 4;
    float4 v = *(const float4*)(in + i);
    ushort4 o;
    o.x = f2bf(v.x); o.y = f2bf(v.y); o.z = f2bf(v.z); o.w = f2bf(v.w);
    *(ushort4*)(out + i) = o;
}

// ---- GEMM: C[M,N] = A[M,Kd] * W[N,Kd]^T  (bf16 in, f32 accum) --------------
// MODE 0: store bf16
// MODE 1: store f32 (acc + resid)
// MODE 2: store bf16 relu(acc + bias)
// MODE 3: store f32 (acc + bias + resid)

template <int MODE>
__global__ __launch_bounds__(256) void gemm_bf16_kernel(
        const unsigned short* __restrict__ A,
        const unsigned short* __restrict__ W,
        unsigned short* __restrict__ outh,
        float* __restrict__ outf,
        const float* __restrict__ bias,
        const float* __restrict__ resid,
        int N, int Kd) {
    __shared__ __align__(16) unsigned short As[2][128 * 40];
    __shared__ __align__(16) unsigned short Bs[2][128 * 40];

    const int tid  = threadIdx.x;
    const int wid  = tid >> 5;
    const int lane = tid & 31;
    const int wm   = wid >> 2;           // 0..1  (M direction, 64 rows each)
    const int wn   = wid & 3;            // 0..3  (N direction, 32 cols each)
    const int m0   = blockIdx.y * 128;
    const int n0   = blockIdx.x * 128;

    const int lm     = lane & 15;
    const int abase  = (lane < 16) ? 0 : 8;    // A-fragment K sub-offset
    const int bbase  = (lane < 16) ? 0 : 8;    // B-fragment K sub-offset

    v8f acc[4][2] = {};

    const int lrow = tid >> 1;                  // 0..127
    const int lcol = (tid & 1) * 16;            // 0 or 16
    const unsigned short* Ag = A + (size_t)(m0 + lrow) * Kd + lcol;
    const unsigned short* Bg = W + (size_t)(n0 + lrow) * Kd + lcol;
    const int lds_off = lrow * 40 + lcol;

#if ASYNC_LDS
    // --- double-buffered async global->LDS pipeline (ASYNCcnt-tracked) ------
    auto issue = [&](int buf, int k0) {
        gld_async16(Ag + k0,     &As[buf][lds_off]);
        gld_async16(Ag + k0 + 8, &As[buf][lds_off + 8]);
        gld_async16(Bg + k0,     &Bs[buf][lds_off]);
        gld_async16(Bg + k0 + 8, &Bs[buf][lds_off + 8]);
    };
    issue(0, 0);
    for (int k0 = 0; k0 < Kd; k0 += 32) {
        const int cur = (k0 >> 5) & 1;
        if (k0 + 32 < Kd) {
            issue(cur ^ 1, k0 + 32);   // prefetch next stage (4 ops in flight)
            wait_async<4>();           // oldest 4 (current buffer) complete
        } else {
            wait_async<0>();
        }
        __syncthreads();               // all waves' tiles visible

        v16bf af[4], bfv[2];
#pragma unroll
        for (int i = 0; i < 4; ++i) {
            const unsigned short* p = &As[cur][(wm * 64 + i * 16 + lm) * 40 + abase];
            af[i] = ld_frag16(p, p + 16);
        }
#pragma unroll
        for (int j = 0; j < 2; ++j) {
            const unsigned short* p = &Bs[cur][(wn * 32 + j * 16 + lm) * 40 + bbase];
            bfv[j] = ld_frag16(p, p + 16);
        }
#pragma unroll
        for (int i = 0; i < 4; ++i)
#pragma unroll
            for (int j = 0; j < 2; ++j)
                acc[i][j] = wmma_bf16(af[i], bfv[j], acc[i][j]);

        __syncthreads();               // reads done before buffer reuse
    }
#else
    // --- synchronous fallback ----------------------------------------------
    for (int k0 = 0; k0 < Kd; k0 += 32) {
        uint4 a0 = *(const uint4*)(Ag + k0);
        uint4 a1 = *(const uint4*)(Ag + k0 + 8);
        uint4 b0 = *(const uint4*)(Bg + k0);
        uint4 b1 = *(const uint4*)(Bg + k0 + 8);
        __syncthreads();
        *(uint4*)(&As[0][lds_off])     = a0;
        *(uint4*)(&As[0][lds_off + 8]) = a1;
        *(uint4*)(&Bs[0][lds_off])     = b0;
        *(uint4*)(&Bs[0][lds_off + 8]) = b1;
        __syncthreads();

        v16bf af[4], bfv[2];
#pragma unroll
        for (int i = 0; i < 4; ++i) {
            const unsigned short* p = &As[0][(wm * 64 + i * 16 + lm) * 40 + abase];
            af[i] = ld_frag16(p, p + 16);
        }
#pragma unroll
        for (int j = 0; j < 2; ++j) {
            const unsigned short* p = &Bs[0][(wn * 32 + j * 16 + lm) * 40 + bbase];
            bfv[j] = ld_frag16(p, p + 16);
        }
#pragma unroll
        for (int i = 0; i < 4; ++i)
#pragma unroll
            for (int j = 0; j < 2; ++j)
                acc[i][j] = wmma_bf16(af[i], bfv[j], acc[i][j]);
    }
#endif

    const int rhalf = (lane < 16) ? 0 : 8;
#pragma unroll
    for (int i = 0; i < 4; ++i) {
#pragma unroll
        for (int j = 0; j < 2; ++j) {
            const int ng = n0 + wn * 32 + j * 16 + lm;
            const int mg = m0 + wm * 64 + i * 16 + rhalf;
#pragma unroll
            for (int e = 0; e < 8; ++e) {
                const size_t idx = (size_t)(mg + e) * N + ng;
                float v = acc[i][j][e];
                if (MODE == 0) {
                    outh[idx] = f2bf(v);
                } else if (MODE == 1) {
                    outf[idx] = v + resid[idx];
                } else if (MODE == 2) {
                    v += bias[ng];
                    outh[idx] = f2bf(v > 0.0f ? v : 0.0f);
                } else {
                    outf[idx] = v + bias[ng] + resid[idx];
                }
            }
        }
    }
}

// ---- Flash attention: per head, streaming softmax ---------------------------
// grid = (S/64, B*H); block = 128 (4 waves x 16 queries).

__global__ __launch_bounds__(128) void attn_kernel(
        const unsigned short* __restrict__ qp,
        const unsigned short* __restrict__ kp,
        const unsigned short* __restrict__ vp,
        const float* __restrict__ mask,
        unsigned short* __restrict__ u_out) {
    __shared__ __align__(16) unsigned short Ps[4][16 * 40];
    __shared__ __align__(16) unsigned short Vs[64 * 40];

    const int tid  = threadIdx.x;
    const int wid  = tid >> 5;
    const int lane = tid & 31;
    const int head = blockIdx.y;
    const int b    = head >> 4;          // head / H
    const int h    = head & 15;          // head % H
    const int cq   = h * KK;             // feature column base for this head
    const int q0w  = blockIdx.x * 64 + wid * 16;
    const int lm   = lane & 15;
    const int rhalf = (lane < 16) ? 0 : 8;
    const int abase = rhalf;             // 0 or 8 (A-fragment K sub-offset)
    const int bbase = (lane < 16) ? 0 : 8;

    // Preload Q fragments (16 queries x 64 features = 2 k-steps).
    v16bf qf[2];
    {
        const size_t rbase = ((size_t)(q0w + lm) * BB + b) * DD + cq;
#pragma unroll
        for (int s = 0; s < 2; ++s) {
            const unsigned short* p = qp + rbase + s * 32 + abase;
            qf[s] = ld_frag16(p, p + 16);
        }
    }

    float mrow[8], lrow[8];
#pragma unroll
    for (int e = 0; e < 8; ++e) { mrow[e] = -3.0e38f; lrow[e] = 0.0f; }
    v8f o[4] = {};

    for (int t0 = 0; t0 < SQ; t0 += 32) {
        // ---- stage V chunk (32 keys x 64 features), transposed to [feat][key]
        {
            const int key = tid >> 2;
            const int seg = (tid & 3) * 16;
            const size_t vr = ((size_t)(t0 + key) * BB + b) * DD + cq + seg;
            union { uint4 u[2]; unsigned short s[16]; } vbuf;
            vbuf.u[0] = *(const uint4*)(vp + vr);
            vbuf.u[1] = *(const uint4*)(vp + vr + 8);
            __syncthreads();
#pragma unroll
            for (int i = 0; i < 16; ++i)
                Vs[(seg + i) * 40 + key] = vbuf.s[i];
            __syncthreads();
        }

        // ---- scores: 16 queries x 32 keys (two 16x16 tiles)
        float xs[2][8];
#pragma unroll
        for (int j = 0; j < 2; ++j) {
            v8f sc = {};
#pragma unroll
            for (int s = 0; s < 2; ++s) {
                const int t = t0 + j * 16 + lm;
                const size_t kr = ((size_t)t * BB + b) * DD + cq + s * 32 + bbase;
                const unsigned short* p = kp + kr;
                v16bf kf = ld_frag16(p, p + 16);
                sc = wmma_bf16(qf[s], kf, sc);
            }
#pragma unroll
            for (int e = 0; e < 8; ++e) {
                const int qrow = q0w + rhalf + e;
                const int t = t0 + j * 16 + lm;
                xs[j][e] = (sc[e] + mask[(size_t)qrow * SQ + t]) * 0.125f;
            }
        }

        // ---- online softmax row statistics (rows live in 16-lane halves)
        float al[8], ps0[8], ps1[8];
#pragma unroll
        for (int e = 0; e < 8; ++e) {
            float rm = fmaxf(xs[0][e], xs[1][e]);
#pragma unroll
            for (int d = 1; d < 16; d <<= 1) rm = fmaxf(rm, __shfl_xor(rm, d));
            const float mn = fmaxf(mrow[e], rm);
            al[e] = __expf(mrow[e] - mn);
            ps0[e] = __expf(xs[0][e] - mn);
            ps1[e] = __expf(xs[1][e] - mn);
            float rs = ps0[e] + ps1[e];
#pragma unroll
            for (int d = 1; d < 16; d <<= 1) rs += __shfl_xor(rs, d);
            lrow[e] = lrow[e] * al[e] + rs;
            mrow[e] = mn;
        }
#pragma unroll
        for (int j2 = 0; j2 < 4; ++j2)
#pragma unroll
            for (int e = 0; e < 8; ++e) o[j2][e] *= al[e];

        // ---- P (16x32) through LDS to reshape C-layout -> A-fragment layout
#pragma unroll
        for (int e = 0; e < 8; ++e) {
            Ps[wid][(rhalf + e) * 40 + lm]      = f2bf(ps0[e]);
            Ps[wid][(rhalf + e) * 40 + 16 + lm] = f2bf(ps1[e]);
        }
        v16bf pf;
        {
            const unsigned short* p = &Ps[wid][lm * 40 + abase];
            pf = ld_frag16(p, p + 16);
        }

        // ---- o += P x V  (4 feature subtiles of 16)
#pragma unroll
        for (int j2 = 0; j2 < 4; ++j2) {
            const unsigned short* p = &Vs[(j2 * 16 + lm) * 40 + bbase];
            v16bf vf = ld_frag16(p, p + 16);
            o[j2] = wmma_bf16(pf, vf, o[j2]);
        }
    }

    // ---- normalize and emit u in [S*B, D] layout (col = h*K + feat)
    float inv[8];
#pragma unroll
    for (int e = 0; e < 8; ++e) inv[e] = 1.0f / lrow[e];
#pragma unroll
    for (int j2 = 0; j2 < 4; ++j2)
#pragma unroll
        for (int e = 0; e < 8; ++e) {
            const int qrow = q0w + rhalf + e;
            const size_t idx = ((size_t)qrow * BB + b) * DD + cq + j2 * 16 + lm;
            u_out[idx] = f2bf(o[j2][e] * inv[e]);
        }
}

// ---- LayerNorm over D=1024; optional bf16 mirror ---------------------------

__global__ __launch_bounds__(256) void ln_kernel(
        const float* __restrict__ in,
        const float* __restrict__ g,
        const float* __restrict__ be,
        float* __restrict__ outf,
        unsigned short* __restrict__ outh) {
    __shared__ float red[16];
    const int row = blockIdx.x;
    const int tid = threadIdx.x;
    const float* r = in + (size_t)row * DD;
    float4 v = *(const float4*)(r + tid * 4);
    float s1 = v.x + v.y + v.z + v.w;
    float s2 = v.x * v.x + v.y * v.y + v.z * v.z + v.w * v.w;
#pragma unroll
    for (int d = 1; d < 32; d <<= 1) {
        s1 += __shfl_xor(s1, d);
        s2 += __shfl_xor(s2, d);
    }
    const int wid = tid >> 5, lane = tid & 31;
    if (lane == 0) { red[wid] = s1; red[8 + wid] = s2; }
    __syncthreads();
    float t1 = 0.0f, t2 = 0.0f;
#pragma unroll
    for (int i = 0; i < 8; ++i) { t1 += red[i]; t2 += red[8 + i]; }
    const float mu  = t1 * (1.0f / DD);
    const float var = t2 * (1.0f / DD) - mu * mu;
    const float rstd = rsqrtf(var + 1e-5f);

    const int c = tid * 4;
    float4 gg = *(const float4*)(g + c);
    float4 bb = *(const float4*)(be + c);
    float4 y;
    y.x = (v.x - mu) * rstd * gg.x + bb.x;
    y.y = (v.y - mu) * rstd * gg.y + bb.y;
    y.z = (v.z - mu) * rstd * gg.z + bb.z;
    y.w = (v.w - mu) * rstd * gg.w + bb.w;
    *(float4*)(outf + (size_t)row * DD + c) = y;
    if (outh) {
        ushort4 o;
        o.x = f2bf(y.x); o.y = f2bf(y.y); o.z = f2bf(y.z); o.w = f2bf(y.w);
        *(ushort4*)(outh + (size_t)row * DD + c) = o;
    }
}

// ---------------------------------------------------------------------------

extern "C" void kernel_launch(void* const* d_in, const int* in_sizes, int n_in,
                              void* d_out, int out_size, void* d_ws, size_t ws_size,
                              hipStream_t stream) {
    (void)in_sizes; (void)n_in; (void)out_size; (void)ws_size;

    const float* x    = (const float*)d_in[0];
    const float* mask = (const float*)d_in[1];
    const float* Wq   = (const float*)d_in[2];
    const float* Wk   = (const float*)d_in[3];
    const float* Wv   = (const float*)d_in[4];
    const float* Wc   = (const float*)d_in[5];
    const float* W1   = (const float*)d_in[6];
    const float* b1   = (const float*)d_in[7];
    const float* W2   = (const float*)d_in[8];
    const float* b2   = (const float*)d_in[9];
    const float* g1   = (const float*)d_in[10];
    const float* be1  = (const float*)d_in[11];
    const float* g2   = (const float*)d_in[12];
    const float* be2  = (const float*)d_in[13];
    float* out = (float*)d_out;

    char* ws = (char*)d_ws;
    // bf16 buffers (bytes)
    const size_t O_XB  = 0;                       // x bf16          16 MB
    const size_t O_WQ  = O_XB  + 16777216;        // Wq bf16          2 MB
    const size_t O_WK  = O_WQ  + 2097152;
    const size_t O_WV  = O_WK  + 2097152;
    const size_t O_WC  = O_WV  + 2097152;
    const size_t O_W1  = O_WC  + 2097152;         // W1 bf16          8 MB
    const size_t O_W2  = O_W1  + 8388608;         // W2 bf16          8 MB
    const size_t O_QP  = O_W2  + 8388608;         // q proj bf16     16 MB
    const size_t O_KP  = O_QP  + 16777216;        // k proj bf16     16 MB
    const size_t O_VP  = O_KP  + 16777216;        // v proj bf16     16 MB
    const size_t O_UA  = O_VP  + 16777216;        // attn out bf16   16 MB
    const size_t O_H   = O_QP;                    // FF hidden bf16  64 MB (aliases q/k/v/u)
    const size_t O_AR  = O_UA  + 16777216;        // attn residual f32  32 MB
    const size_t O_ZP  = O_AR;                    // z pre-LN f32 (aliases O_AR)
    const size_t O_U1  = O_AR  + 33554432;        // LN1 out f32     32 MB
    const size_t O_U1B = O_U1  + 33554432;        // LN1 out bf16    16 MB

    unsigned short* xb  = (unsigned short*)(ws + O_XB);
    unsigned short* wqb = (unsigned short*)(ws + O_WQ);
    unsigned short* wkb = (unsigned short*)(ws + O_WK);
    unsigned short* wvb = (unsigned short*)(ws + O_WV);
    unsigned short* wcb = (unsigned short*)(ws + O_WC);
    unsigned short* w1b = (unsigned short*)(ws + O_W1);
    unsigned short* w2b = (unsigned short*)(ws + O_W2);
    unsigned short* qp  = (unsigned short*)(ws + O_QP);
    unsigned short* kp  = (unsigned short*)(ws + O_KP);
    unsigned short* vp  = (unsigned short*)(ws + O_VP);
    unsigned short* ua  = (unsigned short*)(ws + O_UA);
    unsigned short* hb  = (unsigned short*)(ws + O_H);
    float*          ar  = (float*)(ws + O_AR);
    float*          zp  = (float*)(ws + O_ZP);
    float*          u1  = (float*)(ws + O_U1);
    unsigned short* u1b = (unsigned short*)(ws + O_U1B);

    // 1) convert inputs/weights to bf16
    cvt_bf16_kernel<<<8192, 256, 0, stream>>>(x,  xb);    // 8M elems
    cvt_bf16_kernel<<<1024, 256, 0, stream>>>(Wq, wqb);   // 1M elems each
    cvt_bf16_kernel<<<1024, 256, 0, stream>>>(Wk, wkb);
    cvt_bf16_kernel<<<1024, 256, 0, stream>>>(Wv, wvb);
    cvt_bf16_kernel<<<1024, 256, 0, stream>>>(Wc, wcb);
    cvt_bf16_kernel<<<4096, 256, 0, stream>>>(W1, w1b);   // 4M elems
    cvt_bf16_kernel<<<4096, 256, 0, stream>>>(W2, w2b);

    // 2) q/k/v projections: [8192,1024] = xb @ W^T
    const dim3 gProj(DD / 128, MROWS / 128);              // (8, 64)
    gemm_bf16_kernel<0><<<gProj, 256, 0, stream>>>(xb, wqb, qp, nullptr, nullptr, nullptr, DD, DD);
    gemm_bf16_kernel<0><<<gProj, 256, 0, stream>>>(xb, wkb, kp, nullptr, nullptr, nullptr, DD, DD);
    gemm_bf16_kernel<0><<<gProj, 256, 0, stream>>>(xb, wvb, vp, nullptr, nullptr, nullptr, DD, DD);

    // 3) attention (mask added before 1/sqrt(K) scaling, faithful)
    attn_kernel<<<dim3(SQ / 64, BB * HH), 128, 0, stream>>>(qp, kp, vp, mask, ua);

    // 4) attn output projection + residual: ar = x + ua @ Wc^T
    gemm_bf16_kernel<1><<<gProj, 256, 0, stream>>>(ua, wcb, nullptr, ar, nullptr, x, DD, DD);

    // 5) LN1 -> u1 (f32) + u1b (bf16)
    ln_kernel<<<MROWS, 256, 0, stream>>>(ar, g1, be1, u1, u1b);

    // 6) FFN1: hb = relu(u1b @ W1^T + b1)   [8192, 4096]
    const dim3 gFF1(FFD / 128, MROWS / 128);              // (32, 64)
    gemm_bf16_kernel<2><<<gFF1, 256, 0, stream>>>(u1b, w1b, hb, nullptr, b1, nullptr, FFD, DD);

    // 7) FFN2 + residual: zp = u1 + hb @ W2^T + b2   [8192, 1024]
    gemm_bf16_kernel<3><<<gProj, 256, 0, stream>>>(hb, w2b, nullptr, zp, b2, u1, DD, FFD);

    // 8) LN2 -> d_out
    ln_kernel<<<MROWS, 256, 0, stream>>>(zp, g2, be2, out, nullptr);
}
